// PointNetSetAbstractionCN2Nor_67997922230548
// MI455X (gfx1250) — matrix operand order repaired and checked by reading, not compiled, not run on hardware
//
#include <hip/hip_runtime.h>

typedef _Float16 v16h __attribute__((ext_vector_type(16)));
typedef float    v8f  __attribute__((ext_vector_type(8)));

#define KNN     32
#define BN_EPS  1e-5f

// ---- workspace float layout -------------------------------------------------
#define WS_SLOC    0      // [32]  sum(loc)
#define WS_SSLOC   32     // [32]  sum(loc^2)
#define WS_SFEAT   64     // [32]  sum(feat)
#define WS_SSFEAT  96     // [32]  sum(feat^2)
#define WS_SH2     128    // [64]  sum(h2)
#define WS_SSH2    192    // [64]  sum(h2^2)
#define WS_WFUSED  256    // [32][20] fused BN1-scaled layer-0 weights (col 19 = 0)
#define WS_BSUM    896    // [32]  fused BN1 shift
#define WS_SCALE2  928    // [64]  BN2 scale
#define WS_SHIFT2  992    // [64]  BN2 shift
#define WS_H2MAX   1280   // [N*64] per-(n,o) max_k of pre-BN conv2 output
// h2min follows at WS_H2MAX + N*64

// ---- kernel 0: zero the atomic accumulators (ws is poisoned by harness) -----
__global__ void k0_init(float* wsf) {
    int i = blockIdx.x * blockDim.x + threadIdx.x;
    if (i < 256) wsf[i] = 0.0f;
}

// ---- kernel 1: per-channel sum/sumsq of loc (3->32) and feat (16->32) -------
// Wave = 32 lanes = 32 output channels; rows staged in LDS, broadcast-read.
__global__ __launch_bounds__(256)
void k1_stats1(const float* __restrict__ xyz, const float* __restrict__ pts,
               const int* __restrict__ gidx,
               const float* __restrict__ Wl0, const float* __restrict__ Wf0,
               float* __restrict__ wsf, long long NK) {
    __shared__ float stage[256][20];
    __shared__ float acc[128];

    const int tid  = threadIdx.x;
    const int lane = tid & 31;           // channel index
    const int wbase = tid & ~31;         // this wave's 32 staged rows

    // preload this channel's weights
    float wl[3], wf[16];
#pragma unroll
    for (int i = 0; i < 3; ++i)  wl[i] = Wl0[lane * 3 + i];
#pragma unroll
    for (int i = 0; i < 16; ++i) wf[i] = Wf0[lane * 16 + i];

    float sl = 0.f, ssl = 0.f, sf = 0.f, ssf = 0.f;

    const long long total = (long long)gridDim.x * 256;
    const long long iters = (NK + total - 1) / total;

    if (tid < 128) acc[tid] = 0.f;

    for (long long it = 0; it < iters; ++it) {
        long long r = (long long)blockIdx.x * 256 + it * total + tid;
        __syncthreads();
        if (r < NK) {
            int n  = (int)(r >> 5);                 // K = 32
            int gi = gidx[r];
            stage[tid][0] = xyz[gi * 3 + 0] - xyz[n * 3 + 0];
            stage[tid][1] = xyz[gi * 3 + 1] - xyz[n * 3 + 1];
            stage[tid][2] = xyz[gi * 3 + 2] - xyz[n * 3 + 2];
            const float4* p4 = (const float4*)pts;
#pragma unroll
            for (int w = 0; w < 4; ++w) {
                float4 q = p4[(long long)gi * 4 + w];
                stage[tid][3 + 4 * w + 0] = q.x;
                stage[tid][3 + 4 * w + 1] = q.y;
                stage[tid][3 + 4 * w + 2] = q.z;
                stage[tid][3 + 4 * w + 3] = q.w;
            }
        } else {
#pragma unroll
            for (int i = 0; i < 19; ++i) stage[tid][i] = 0.f;   // zero rows: no-op for sums
        }
        __syncthreads();
#pragma unroll 4
        for (int j = 0; j < 32; ++j) {
            const float* u = stage[wbase + j];
            float loc = fmaf(u[0], wl[0], fmaf(u[1], wl[1], u[2] * wl[2]));
            float ft = 0.f;
#pragma unroll
            for (int i = 0; i < 16; ++i) ft = fmaf(u[3 + i], wf[i], ft);
            sl += loc;  ssl = fmaf(loc, loc, ssl);
            sf += ft;   ssf = fmaf(ft, ft, ssf);
        }
    }
    __syncthreads();
    atomicAdd(&acc[lane],      sl);
    atomicAdd(&acc[32 + lane], ssl);
    atomicAdd(&acc[64 + lane], sf);
    atomicAdd(&acc[96 + lane], ssf);
    __syncthreads();
    if (tid < 128) atomicAdd(&wsf[WS_SLOC + tid], acc[tid]);
}

// ---- kernel 2: finalize BN1, build fused 19->32 weights ---------------------
__global__ void k2_fuse(const float* __restrict__ Wl0, const float* __restrict__ gl0,
                        const float* __restrict__ bl0, const float* __restrict__ Wf0,
                        const float* __restrict__ gf0, const float* __restrict__ bf0,
                        float* __restrict__ wsf, float invNK) {
    int c = threadIdx.x;
    if (c >= 32) return;
    float ml = wsf[WS_SLOC + c]  * invNK;
    float vl = wsf[WS_SSLOC + c] * invNK - ml * ml;
    float al = gl0[c] * rsqrtf(vl + BN_EPS);
    float bl = bl0[c] - ml * al;
    float mf = wsf[WS_SFEAT + c]  * invNK;
    float vf = wsf[WS_SSFEAT + c] * invNK - mf * mf;
    float af = gf0[c] * rsqrtf(vf + BN_EPS);
    float bf = bf0[c] - mf * af;
    wsf[WS_BSUM + c] = bl + bf;
    float* W = &wsf[WS_WFUSED + c * 20];
    for (int i = 0; i < 3;  ++i) W[i]     = al * Wl0[c * 3 + i];
    for (int i = 0; i < 16; ++i) W[3 + i] = af * Wf0[c * 16 + i];
    W[19] = 0.f;
}

// ---- kernel 3: main WMMA pass -----------------------------------------------
// One wave per point: layer0 (19->32, fused BN1) + ReLU via WMMA, LDS transpose,
// layer1 (32->64) via WMMA, accumulate h2 stats + per-point K-max/min.
__global__ __launch_bounds__(256)
void k3_main(const float* __restrict__ xyz, const float* __restrict__ pts,
             const int* __restrict__ gidx, const float* __restrict__ wsf,
             const float* __restrict__ W1, const float* __restrict__ b1,
             float* __restrict__ h2max, float* __restrict__ h2min,
             float* __restrict__ sh2, float* __restrict__ ssh2, int Npts) {
    __shared__ _Float16 ht[8][32][33];     // per-wave 32x32 h tile (padded)
    __shared__ float    sacc[128];

    const int tid    = threadIdx.x;
    const int lane   = tid & 31;
    const int winblk = tid >> 5;
    const int col    = lane & 15;
    const int hi     = lane >> 4;          // 0: first K-half, 1: second K-half
    const int wid    = (blockIdx.x * blockDim.x + tid) >> 5;
    const int nWaves = (gridDim.x * blockDim.x) >> 5;

    if (tid < 128) sacc[tid] = 0.f;

    // B fragments, layer 0: Wfused^T [19(pad32) x 16] per output tile t
    const float* WF = &wsf[WS_WFUSED];
    v16h b0[2];
#pragma unroll
    for (int t = 0; t < 2; ++t) {
        int c = t * 16 + col;
#pragma unroll
        for (int e = 0; e < 16; ++e) {
            int kk = hi * 8 + (e < 8 ? e : e + 8);
            b0[t][e] = (kk < 20) ? (_Float16)WF[c * 20 + kk] : (_Float16)0.f;
        }
    }
    // B fragments, layer 1: W1^T [32 x 16] per output tile t (K = 32 exact)
    v16h b1f[4];
#pragma unroll
    for (int t = 0; t < 4; ++t) {
        int c = t * 16 + col;
#pragma unroll
        for (int e = 0; e < 16; ++e)
            b1f[t][e] = (_Float16)W1[c * 32 + hi * 8 + (e < 8 ? e : e + 8)];
    }
    float bs0[2], bb[4];
    bs0[0] = wsf[WS_BSUM + col];  bs0[1] = wsf[WS_BSUM + 16 + col];
#pragma unroll
    for (int t = 0; t < 4; ++t) bb[t] = b1[t * 16 + col];

    float s[4] = {0.f, 0.f, 0.f, 0.f}, ss[4] = {0.f, 0.f, 0.f, 0.f};
    _Float16 (*myt)[33] = ht[winblk];

    const int iters = (Npts + nWaves - 1) / nWaves;
    for (int it = 0; it < iters; ++it) {
        const int p = wid + it * nWaves;
        const bool act = p < Npts;
        __syncthreads();                                   // protect prev-iter LDS reads
        if (act) {
            if (it + 1 < iters)
                __builtin_prefetch(gidx + (long long)(p + nWaves) * KNN, 0, 1);
            float cx = xyz[p * 3 + 0], cy = xyz[p * 3 + 1], cz = xyz[p * 3 + 2];
            // ---- layer 0: build A (u = [dx, points], K padded to 32) ----
            v16h a0[2];
#pragma unroll
            for (int t2 = 0; t2 < 2; ++t2) {
                int k  = t2 * 16 + col;                    // A row = lane&15
                int gi = gidx[(long long)p * KNN + k];
                float u[20];
                u[0] = xyz[gi * 3 + 0] - cx;
                u[1] = xyz[gi * 3 + 1] - cy;
                u[2] = xyz[gi * 3 + 2] - cz;
                const float4* p4 = (const float4*)pts;
#pragma unroll
                for (int w = 0; w < 4; ++w) {
                    float4 q = p4[(long long)gi * 4 + w];
                    u[3 + 4 * w + 0] = q.x; u[3 + 4 * w + 1] = q.y;
                    u[3 + 4 * w + 2] = q.z; u[3 + 4 * w + 3] = q.w;
                }
                u[19] = 0.f;
                int kb = hi * 8;                           // lane K-half base
#pragma unroll
                for (int j = 0; j < 8; ++j) {
                    int k0 = kb + j, k1 = kb + 16 + j;     // ISA A-layout: e<8 -> K=e, e>=8 -> K=e+8
                    a0[t2][j]     = (k0 < 20) ? (_Float16)u[k0] : (_Float16)0.f;
                    a0[t2][8 + j] = (k1 < 20) ? (_Float16)u[k1] : (_Float16)0.f;
                }
            }
            // ---- layer 0 WMMA + ReLU -> LDS h tile ----
#pragma unroll
            for (int t2 = 0; t2 < 2; ++t2) {
#pragma unroll
                for (int t = 0; t < 2; ++t) {
                    v8f c0;
#pragma unroll
                    for (int v = 0; v < 8; ++v) c0[v] = bs0[t];   // bias in accumulator
                    v8f d = __builtin_amdgcn_wmma_f32_16x16x32_f16(
                        false, a0[t2], false, b0[t], (short)0, c0, false, false);
                    int c  = t * 16 + col;                 // D: col = lane&15
                    int rb = t2 * 16 + hi * 8;             // D: rows v (+8 for lanes>=16)
#pragma unroll
                    for (int v = 0; v < 8; ++v)
                        myt[rb + v][c] = (_Float16)fmaxf(d[v], 0.f);
                }
            }
        }
        __syncthreads();                                   // publish h tile
        if (act) {
            // ---- layer 1: read A from LDS in ISA A-layout ----
            v16h a1[2];
            int lo = hi * 8;
#pragma unroll
            for (int t2 = 0; t2 < 2; ++t2) {
                int r = t2 * 16 + col;
#pragma unroll
                for (int j = 0; j < 8; ++j) {
                    a1[t2][j]     = myt[r][lo + j];
                    a1[t2][8 + j] = myt[r][lo + 16 + j];
                }
            }
            float mx[4], mn[4];
#pragma unroll
            for (int t = 0; t < 4; ++t) { mx[t] = -3.4e38f; mn[t] = 3.4e38f; }
#pragma unroll
            for (int t2 = 0; t2 < 2; ++t2) {
#pragma unroll
                for (int t = 0; t < 4; ++t) {
                    v8f cc;
#pragma unroll
                    for (int v = 0; v < 8; ++v) cc[v] = bb[t];    // conv bias b1
                    v8f d = __builtin_amdgcn_wmma_f32_16x16x32_f16(
                        false, a1[t2], false, b1f[t], (short)0, cc, false, false);
#pragma unroll
                    for (int v = 0; v < 8; ++v) {
                        float x = d[v];
                        s[t] += x; ss[t] = fmaf(x, x, ss[t]);
                        mx[t] = fmaxf(mx[t], x); mn[t] = fminf(mn[t], x);
                    }
                }
            }
            // combine the two K-halves held by lane pairs (L, L+16)
#pragma unroll
            for (int t = 0; t < 4; ++t) {
                mx[t] = fmaxf(mx[t], __shfl_xor(mx[t], 16));
                mn[t] = fminf(mn[t], __shfl_xor(mn[t], 16));
            }
            if (lane < 16) {
#pragma unroll
                for (int t = 0; t < 4; ++t) {
                    h2max[(long long)p * 64 + t * 16 + lane] = mx[t];
                    h2min[(long long)p * 64 + t * 16 + lane] = mn[t];
                }
            }
        }
    }
    __syncthreads();
#pragma unroll
    for (int t = 0; t < 4; ++t) {
        atomicAdd(&sacc[t * 16 + col],      s[t]);
        atomicAdd(&sacc[64 + t * 16 + col], ss[t]);
    }
    __syncthreads();
    if (tid < 64)  atomicAdd(&sh2[tid],       sacc[tid]);
    else if (tid < 128) atomicAdd(&ssh2[tid - 64], sacc[tid]);
}

// ---- kernel 4: finalize BN2 -------------------------------------------------
__global__ void k4_bn2(const float* __restrict__ g1, const float* __restrict__ beta1,
                       float* __restrict__ wsf, float invNK) {
    int o = threadIdx.x;
    if (o >= 64) return;
    float m = wsf[WS_SH2 + o]  * invNK;
    float v = wsf[WS_SSH2 + o] * invNK - m * m;
    float a = g1[o] * rsqrtf(v + BN_EPS);
    wsf[WS_SCALE2 + o] = a;
    wsf[WS_SHIFT2 + o] = beta1[o] - m * a;
}

// ---- kernel 5: epilogue: pick max/min per BN2 sign, apply BN2 + ReLU --------
__global__ void k5_out(const float* __restrict__ wsf, const float* __restrict__ h2max,
                       const float* __restrict__ h2min, float* __restrict__ out,
                       long long total) {
    long long i = (long long)blockIdx.x * blockDim.x + threadIdx.x;
    long long stride = (long long)gridDim.x * blockDim.x;
    for (; i < total; i += stride) {
        int o = (int)(i & 63);
        float a = wsf[WS_SCALE2 + o];
        float x = (a >= 0.f) ? h2max[i] : h2min[i];   // max commutes with monotone affine+ReLU
        out[i] = fmaxf(fmaf(a, x, wsf[WS_SHIFT2 + o]), 0.f);
    }
}

extern "C" void kernel_launch(void* const* d_in, const int* in_sizes, int n_in,
                              void* d_out, int out_size, void* d_ws, size_t ws_size,
                              hipStream_t stream) {
    const float* xyz   = (const float*)d_in[0];
    const float* pts   = (const float*)d_in[1];
    const int*   gidx  = (const int*)  d_in[2];
    const float* Wl0   = (const float*)d_in[3];
    const float* gl0   = (const float*)d_in[4];
    const float* bl0   = (const float*)d_in[5];
    const float* Wf0   = (const float*)d_in[6];
    const float* gf0   = (const float*)d_in[7];
    const float* bf0   = (const float*)d_in[8];
    const float* W1    = (const float*)d_in[9];
    const float* b1    = (const float*)d_in[10];
    const float* g1    = (const float*)d_in[11];
    const float* beta1 = (const float*)d_in[12];

    const int N = in_sizes[0] / 3;
    const long long NK = (long long)N * KNN;
    const float invNK = 1.0f / (float)NK;

    float* wsf   = (float*)d_ws;
    float* h2max = wsf + WS_H2MAX;
    float* h2min = h2max + (long long)N * 64;

    k0_init<<<1, 256, 0, stream>>>(wsf);

    int g1b = (int)((NK + 255) / 256); if (g1b > 1024) g1b = 1024;
    k1_stats1<<<g1b, 256, 0, stream>>>(xyz, pts, gidx, Wl0, Wf0, wsf, NK);

    k2_fuse<<<1, 32, 0, stream>>>(Wl0, gl0, bl0, Wf0, gf0, bf0, wsf, invNK);

    int g3b = (N + 7) / 8; if (g3b > 2048) g3b = 2048;
    k3_main<<<g3b, 256, 0, stream>>>(xyz, pts, gidx, wsf, W1, b1,
                                     h2max, h2min,
                                     wsf + WS_SH2, wsf + WS_SSH2, N);

    k4_bn2<<<1, 64, 0, stream>>>(g1, beta1, wsf, invNK);

    long long total = (long long)N * 64;
    int g5b = (int)((total + 255) / 256); if (g5b > 4096) g5b = 4096;
    k5_out<<<g5b, 256, 0, stream>>>(wsf, h2max, h2min, (float*)d_out, total);
}
